// Involution2D_8847632630036
// MI455X (gfx1250) — compile-verified, matrix operand order checked by reading, not run on hardware
//
#include <hip/hip_runtime.h>
#include <hip/hip_bf16.h>

// Involution2D: CH=256, K=7, P=3, G=16, RED=4, BS=4, H=W=64
// Phase 0: weights -> bf16 WMMA-fragment layout (once)
// Phase 1: two chained GEMMs on v_wmma_f32_16x16x32_bf16 -> per-pixel kernels
// Phase 2: VALU gather apply (coalesced, L2 resident)

typedef __attribute__((ext_vector_type(16))) __bf16 v16bf;
typedef __attribute__((ext_vector_type(8)))  float  v8f;

#define NPIX   16384   // BS*H*W
#define TILEPX 16

// ws layout (bytes):
//   [0            .. 32768)      wrb : GEMM1 B frags, 4 nt * 8 kk * 32 lane * 16 bf16
//   [32768        .. 133120)     wsb : GEMM2 B frags, 49 nt * 2 kk * 32 lane * 16 bf16
//   [133120       .. +51.4MB)    kern: (784, 16384) f32, [o][pixel]
#define WRB_OFF  0
#define WSB_OFF  32768
#define KERN_OFF 133120
#define WRB_N    16384
#define WSB_N    50176

__device__ __forceinline__ unsigned short f2bf(float f) {
    unsigned int u = __float_as_uint(f);
    unsigned int r = u + 0x7FFFu + ((u >> 16) & 1u);   // RNE
    return (unsigned short)(r >> 16);
}
// ISA 7.12.2 16-bit A/B fragment: element pair pr (0..7), lane half hi.
__device__ __forceinline__ int koff(int pr, int hi) {
    return ((pr & 4) ? 16 : 0) + 8 * hi + ((pr & 3) << 1);
}
// inverse map: K-within-32 -> (hi, element index e)
__device__ __forceinline__ void kinv(int kl, int& h2, int& e) {
    h2 = (kl >> 3) & 1;
    int t = kl - 8 * h2;                    // {0..7} U {16..23}
    int pr = (t >= 16) ? (4 + ((t - 16) >> 1)) : (t >> 1);
    e = 2 * pr + (kl & 1);
}

union frag_u { v16bf v; uint4 q[2]; };

// ---------------- Phase 0: weights -> bf16 fragment order --------------------
__global__ __launch_bounds__(256) void prep_weights(
    const float* __restrict__ wr,   // (64,256)
    const float* __restrict__ wsp,  // (784,64)
    unsigned short* __restrict__ wrb,
    unsigned short* __restrict__ wsb)
{
    int idx = blockIdx.x * 256 + threadIdx.x;
    if (idx < WRB_N) {
        // idx = ((nt*8 + kk)*32 + lane)*16 + e
        int e = idx & 15, lane = (idx >> 4) & 31, kk = (idx >> 9) & 7, nt = idx >> 12;
        int n = nt * 16 + (lane & 15), hi = lane >> 4;
        int k = kk * 32 + koff(e >> 1, hi) + (e & 1);
        wrb[idx] = f2bf(wr[n * 256 + k]);
    } else if (idx < WRB_N + WSB_N) {
        int j = idx - WRB_N;                 // j = ((nt*2 + kk)*32 + lane)*16 + e
        int e = j & 15, lane = (j >> 4) & 31, t2 = j >> 9;
        int kk = t2 & 1, nt = t2 >> 1;
        int n = nt * 16 + (lane & 15), hi = lane >> 4;
        int k = kk * 32 + koff(e >> 1, hi) + (e & 1);
        wsb[j] = f2bf(wsp[n * 64 + k]);
    }
}

// ---------------- Phase 1: per-pixel kernel generation -----------------------
__global__ __launch_bounds__(256) void involution_kern_gen(
    const float* __restrict__ x,          // (4,256,64,64)
    const float* __restrict__ br,         // (64)
    const float* __restrict__ bsp,        // (784)
    const unsigned short* __restrict__ wrb,
    const unsigned short* __restrict__ wsb,
    float* __restrict__ kern)             // (784, 16384)
{
    __shared__ __align__(16) unsigned short lds_x[8 * 32 * 16]; // frag order, 8 ksteps
    __shared__ __align__(16) unsigned short lds_t[2 * 32 * 16]; // frag order, 2 ksteps

    const int tid   = threadIdx.x;
    const int lane  = tid & 31;
    const int wave  = tid >> 5;
    const int m16   = lane & 15;
    const int hi    = lane >> 4;
    const int pbase = blockIdx.x * TILEPX;

    // ---- stage x tile (16 px x 256 ch) -> LDS bf16 in fragment order
    for (int it = 0; it < 16; ++it) {
        int id = it * 256 + tid;
        int c  = id >> 4;                 // channel 0..255 (= K of GEMM1)
        int m  = id & 15;                 // pixel-in-tile (= M)
        int pg = pbase + m;
        int b  = pg >> 12;
        int pi = pg & 4095;
        unsigned short v = f2bf(x[(b * 256 + c) * 4096 + pi]);
        int kk = c >> 5, h2, e;
        kinv(c & 31, h2, e);
        lds_x[(kk * 32 + h2 * 16 + m) * 16 + e] = v;
    }
    __syncthreads();

    // ---- GEMM1: t(16x64) = A(16x256) x w_reduce^T ; waves 0..3 own N-tiles
    if (wave < 4) {
        v8f acc = {};
        const int ng = wave * 16 + m16;            // reduced channel 0..63
        const uint4* wf = (const uint4*)wrb;
#pragma unroll 2
        for (int kk = 0; kk < 8; ++kk) {
            frag_u a, bfr;
            const uint4* lp = (const uint4*)&lds_x[(kk * 32 + lane) * 16];
            a.q[0] = lp[0]; a.q[1] = lp[1];
            const uint4* gp = wf + (size_t)((wave * 8 + kk) * 32 + lane) * 2;
            bfr.q[0] = gp[0]; bfr.q[1] = gp[1];
            acc = __builtin_amdgcn_wmma_f32_16x16x32_bf16(
                      false, a.v, false, bfr.v, (short)0, acc, false, false);
        }
        const float bias = br[ng];
        // scatter t into lds_t fragment order: (m = r + 8*hi, k = ng)
        int kk2 = ng >> 5, h2, e;
        kinv(ng & 31, h2, e);
#pragma unroll
        for (int r = 0; r < 8; ++r) {
            lds_t[(kk2 * 32 + h2 * 16 + (r + 8 * hi)) * 16 + e] = f2bf(acc[r] + bias);
        }
    }
    __syncthreads();

    // ---- GEMM2: kern(16x784) = t(16x64) x w_span^T ; 49 N-tiles over 8 waves
    for (int nt = wave; nt < 49; nt += 8) {
        v8f acc = {};
        const int og = nt * 16 + m16;              // span output 0..783
        const uint4* wf = (const uint4*)wsb;
#pragma unroll
        for (int kk = 0; kk < 2; ++kk) {
            frag_u a, bfr;
            const uint4* lp = (const uint4*)&lds_t[(kk * 32 + lane) * 16];
            a.q[0] = lp[0]; a.q[1] = lp[1];
            const uint4* gp = wf + (size_t)((nt * 2 + kk) * 32 + lane) * 2;
            bfr.q[0] = gp[0]; bfr.q[1] = gp[1];
            acc = __builtin_amdgcn_wmma_f32_16x16x32_bf16(
                      false, a.v, false, bfr.v, (short)0, acc, false, false);
        }
        const float bias = bsp[og];
#pragma unroll
        for (int r = 0; r < 8; ++r) {
            kern[og * NPIX + pbase + (r + 8 * hi)] = acc[r] + bias;
        }
    }
}

// ---------------- Phase 2: involution apply (gather + 49 FMAs) ---------------
__global__ __launch_bounds__(256) void involution_apply(
    const float* __restrict__ x,
    const float* __restrict__ kern,   // (784, 16384)
    float* __restrict__ out)          // (4,256,64,64)
{
    const int idx = blockIdx.x * 256 + threadIdx.x;   // ((b*256+c)*64+h)*64+w
    const int w = idx & 63;
    const int h = (idx >> 6) & 63;
    const int c = (idx >> 12) & 255;
    const int b = idx >> 20;
    const int g = c >> 4;

    const int pglob = b * 4096 + h * 64 + w;
    const float* kp = kern + (g * 49) * NPIX + pglob;
    const float* xp = x + (b * 256 + c) * 4096;

    float acc = 0.f;
#pragma unroll
    for (int i = 0; i < 7; ++i) {
        const int hh = h + i - 3;
        const bool hv = (unsigned)hh < 64u;
#pragma unroll
        for (int j = 0; j < 7; ++j) {
            const int ww = w + j - 3;
            const float kv = kp[(i * 7 + j) * NPIX];
            const float xv = (hv && (unsigned)ww < 64u) ? xp[hh * 64 + ww] : 0.f;
            acc = fmaf(kv, xv, acc);
        }
    }
    out[idx] = acc;
}

extern "C" void kernel_launch(void* const* d_in, const int* in_sizes, int n_in,
                              void* d_out, int out_size, void* d_ws, size_t ws_size,
                              hipStream_t stream) {
    const float* x   = (const float*)d_in[0];
    const float* wr  = (const float*)d_in[1];
    const float* br  = (const float*)d_in[2];
    const float* wsp = (const float*)d_in[3];
    const float* bsp = (const float*)d_in[4];

    char* ws = (char*)d_ws;
    unsigned short* wrb  = (unsigned short*)(ws + WRB_OFF);
    unsigned short* wsb  = (unsigned short*)(ws + WSB_OFF);
    float*          kern = (float*)(ws + KERN_OFF);
    float*          out  = (float*)d_out;

    prep_weights<<<(WRB_N + WSB_N + 255) / 256, 256, 0, stream>>>(wr, wsp, wrb, wsb);
    involution_kern_gen<<<NPIX / TILEPX, 256, 0, stream>>>(x, br, bsp, wrb, wsb, kern);
    involution_apply<<<(4 * 256 * 64 * 64) / 256, 256, 0, stream>>>(x, kern, out);
}
